// PointNetAutoEncoder_65927747994189
// MI455X (gfx1250) — matrix operand order repaired and checked by reading, not compiled, not run on hardware
//
#include <hip/hip_runtime.h>
#include <stdint.h>

// ---------------------------------------------------------------------------
// PointNet autoencoder for MI455X (gfx1250, wave32, WMMA bf16->f32).
// Fused per-point MLP head kernel: 5 conv layers + LN + ReLU + max-over-N,
// activations and weight tiles LDS-resident, matmuls via
// v_wmma_f32_16x16x32_bf16, 1024-layer weight chunks staged with the
// Tensor Data Mover (tensor_load_to_lds + s_wait_tensorcnt).
// ---------------------------------------------------------------------------

constexpr int ASTR  = 136;  // LDS activation row stride (bf16), 16B-aligned rows
constexpr int ZSTR  = 132;  // LDS z row stride (f32)
constexpr int WTSTR = 72;   // transposed weight stride [col][k] (bf16), 16B-aligned

typedef __attribute__((ext_vector_type(16))) __bf16          v16bf;
typedef __attribute__((ext_vector_type(16))) unsigned short  v16u;
typedef __attribute__((ext_vector_type(8)))  float           v8f;
typedef __attribute__((ext_vector_type(4)))  unsigned int    v4u;
typedef __attribute__((ext_vector_type(8)))  int             v8i;
typedef __attribute__((ext_vector_type(4)))  int             v4i;

#if defined(__has_builtin)
#if __has_builtin(__builtin_amdgcn_tensor_load_to_lds) && \
    __has_builtin(__builtin_amdgcn_s_wait_tensorcnt)
#define HAVE_TDM 1
#endif
#endif

__device__ inline unsigned short f2bf(float f) {
  unsigned int u = __float_as_uint(f);
  u += 0x7FFFu + ((u >> 16) & 1u);          // round-to-nearest-even
  return (unsigned short)(u >> 16);
}

__device__ inline v16bf bitcast16(v16u u) {
  union { v16u u; v16bf b; } c; c.u = u; return c.b;
}

// A-matrix 16x32 bf16 from LDS (row-major, stride ASTR).  base = &tile[row0][k0].
__device__ inline v16bf load_A_lds(const unsigned short* base, int lane) {
  int row = lane & 15;
  int kb  = (lane & 16) ? 16 : 0;           // lanes 16-31 hold K 16..31
  const unsigned short* p = base + row * ASTR + kb;
  v16u a;
#pragma unroll
  for (int vg = 0; vg < 4; ++vg) {
    a[2*vg]   = p[2*vg];
    a[2*vg+1] = p[2*vg+1];
    a[2*vg+8] = p[8 + 2*vg];
    a[2*vg+9] = p[8 + 2*vg+1];
  }
  return bitcast16(a);
}

// A-matrix 16x32 from global f32 (convert to bf16 on load).  base = &A[row0][0].
__device__ inline v16bf load_A_gf32(const float* base, int lda, int k0, int lane) {
  int row = lane & 15;
  int kb  = (lane & 16) ? 16 : 0;
  const float* p = base + (size_t)row * lda + k0 + kb;
  v16u a;
#pragma unroll
  for (int vg = 0; vg < 4; ++vg) {
    a[2*vg]   = f2bf(p[2*vg]);
    a[2*vg+1] = f2bf(p[2*vg+1]);
    a[2*vg+8] = f2bf(p[8 + 2*vg]);
    a[2*vg+9] = f2bf(p[8 + 2*vg+1]);
  }
  return bitcast16(a);
}

// B-matrix 32x16 from LDS, TRANSPOSED layout wT[col][k]: per lane the 16
// K-values are contiguous -> vectorizes to ds_load_b128.  base = &wT[col0][k0].
__device__ inline v16bf load_B_ldsT(const unsigned short* base, int lane) {
  int col = lane & 15;
  int kb  = (lane & 16) ? 16 : 0;           // lanes 16-31 hold K 16..31
  const unsigned short* p = base + col * WTSTR + kb;
  v16u b;
#pragma unroll
  for (int j = 0; j < 16; ++j) b[j] = p[j];
  return bitcast16(b);
}

// B-matrix 32x16 from LDS, row-major K x 128 chunk (TDM-staged).
__device__ inline v16bf load_B_ldsRM(const unsigned short* base, int lane) {
  int col = lane & 15;
  int kb  = (lane & 16) ? 16 : 0;
  const unsigned short* p = base + (size_t)kb * 128 + col;
  v16u b;
#pragma unroll
  for (int vg = 0; vg < 8; ++vg) {
    b[2*vg]   = p[(2*vg)   * 128];
    b[2*vg+1] = p[(2*vg+1) * 128];
  }
  return bitcast16(b);
}

// B-matrix 32x16 bf16 from global (row-major K x N).  W = &W[k0][col0].
__device__ inline v16bf load_B_g(const unsigned short* W, int ldb, int lane) {
  int col = lane & 15;
  int kb  = (lane & 16) ? 16 : 0;
  const unsigned short* p = W + (size_t)kb * ldb + col;
  v16u b;
#pragma unroll
  for (int vg = 0; vg < 8; ++vg) {
    b[2*vg]   = p[(size_t)(2*vg)   * ldb];
    b[2*vg+1] = p[(size_t)(2*vg+1) * ldb];
  }
  return bitcast16(b);
}

// ---------------------------------------------------------------------------
// TDM stage of one 128(K) x 128(N) bf16 weight chunk, row stride 1024 elems,
// into LDS row-major (contiguous 32KB tile).  Issued by wave 0 only.
// 6-arg builtin variant (clang-23 / therock-10.0): extra zero int32x8 group.
// ---------------------------------------------------------------------------
__device__ inline void stage_w5_chunk(const unsigned short* gsrc,
                                      unsigned short* wstage,
                                      int wid, int tid) {
#if defined(HAVE_TDM)
  if (wid == 0) {
    unsigned long long ga = (unsigned long long)(uintptr_t)gsrc;
    unsigned ldsoff = (unsigned)(uintptr_t)wstage;     // LDS byte offset
    // D# group 0: count=1 | lds_addr | global_addr[56:0] | type=2
    v4u g0 = { 1u, ldsoff, (unsigned)ga,
               (unsigned)((ga >> 32) & 0x01FFFFFFu) | (2u << 30) };
    // D# group 1: data_size=2B; tensor_dim0=1024; tensor_dim1=128;
    //             tile_dim0=128; tile_dim1=128; tensor_dim0_stride=1024
    v8i g1 = { (int)(1u << 16),                         // wg_mask=0, data_size=1(2B)
               (int)((1024u & 0xFFFFu) << 16),          // tensor_dim0[15:0]
               (int)(((1024u >> 16) & 0xFFFFu) | ((128u & 0xFFFFu) << 16)),
               (int)(((128u >> 16) & 0xFFFFu) | (128u << 16)),  // tile_dim0=128
               (int)(128u),                             // tile_dim1=128, tile_dim2=0
               (int)1024,                               // stride0 low32
               0, 0 };
    v4i gz4 = { 0, 0, 0, 0 };                           // groups 2/3: 2D tensor
    v8i gz8 = { 0, 0, 0, 0, 0, 0, 0, 0 };
    __builtin_amdgcn_tensor_load_to_lds(g0, g1, gz4, gz4, gz8, 0);
    __builtin_amdgcn_s_wait_tensorcnt(0);
  }
  (void)tid;
  __syncthreads();
#else
  for (int i = tid; i < 128 * 128; i += 256) {
    int k = i >> 7, c = i & 127;
    wstage[i] = gsrc[(size_t)k * 1024 + c];
  }
  (void)wid;
  __syncthreads();
#endif
}

// ---------------------------------------------------------------------------
// Fused conv layer: stage W (KxNOUT) transposed into LDS, then
// z = A(32xK) @ W + b ; LN over NOUT ; ReLU -> abuf.
// 8 waves split 2x(NOUT/16) 16x16 WMMA tiles.
// ---------------------------------------------------------------------------
template <int K, int NOUT>
__device__ void conv_layer(unsigned short (&abuf)[32][ASTR], float (&zbuf)[32][ZSTR],
                           unsigned short* __restrict__ wT,
                           const unsigned short* __restrict__ W,
                           const float* __restrict__ bias,
                           const float* __restrict__ g,
                           const float* __restrict__ be,
                           int wid, int lane, int tid) {
  // stage weights transposed: wT[col][k]  (global reads coalesced over cols)
  for (int i = tid; i < K * NOUT; i += 256) {
    int k = i / NOUT, c = i - k * NOUT;
    wT[c * WTSTR + k] = W[i];
  }
  __syncthreads();

  constexpr int NCT = NOUT / 16;
  for (int t = wid; t < 2 * NCT; t += 8) {
    int rt = t / NCT, ct = t % NCT;
    v8f acc = {};
#pragma unroll
    for (int k0 = 0; k0 < K; k0 += 32) {
      v16bf A = load_A_lds(&abuf[rt * 16][k0], lane);
      v16bf B = load_B_ldsT(wT + (ct * 16) * WTSTR + k0, lane);
      acc = __builtin_amdgcn_wmma_f32_16x16x32_bf16(false, A, false, B, (short)0,
                                                    acc, false, false);
    }
    int col   = ct * 16 + (lane & 15);
    int rbase = rt * 16 + ((lane & 16) ? 8 : 0);
    float bv  = bias[col];
#pragma unroll
    for (int r = 0; r < 8; ++r) zbuf[rbase + r][col] = acc[r] + bv;
  }
  __syncthreads();
  // LayerNorm + ReLU, one wave per 4 rows, lane-parallel reduction
  for (int row = wid; row < 32; row += 8) {
    float s = 0.f, ss = 0.f;
    for (int c = lane; c < NOUT; c += 32) { float z = zbuf[row][c]; s += z; ss += z * z; }
#pragma unroll
    for (int off = 16; off > 0; off >>= 1) {
      s  += __shfl_xor(s,  off, 32);
      ss += __shfl_xor(ss, off, 32);
    }
    float mean = s * (1.f / NOUT);
    float var  = ss * (1.f / NOUT) - mean * mean;
    float rstd = rsqrtf(var + 1e-3f);
    for (int c = lane; c < NOUT; c += 32) {
      float z = (zbuf[row][c] - mean) * rstd * g[c] + be[c];
      abuf[row][c] = f2bf(z > 0.f ? z : 0.f);
    }
  }
  __syncthreads();
}

// ---------------------------------------------------------------------------
// Final 128->1024 layer.  Two-pass recompute with TDM-staged weight chunks:
// pass1 accumulates per-row sum/sumsq (LDS float atomics), pass2 recomputes z,
// applies LN+ReLU and max-reduces each channel over the 32 points -> global
// uint-ordered atomicMax (post-ReLU values >= 0, buffer pre-zeroed).
// ---------------------------------------------------------------------------
__device__ void conv_final(unsigned short (&abuf)[32][ASTR], float (&zbuf)[32][ZSTR],
                           unsigned short* __restrict__ wstage,
                           const unsigned short* __restrict__ W,
                           const float* __restrict__ bias,
                           const float* __restrict__ g,
                           const float* __restrict__ be,
                           float* __restrict__ outmax,  // pre-offset by group*1024
                           int wid, int lane, int tid,
                           float* rmean, float* rrstd, float* rs, float* rss) {
  if (tid < 32) { rs[tid] = 0.f; rss[tid] = 0.f; }
  __syncthreads();
  // pass 1: statistics over all 1024 channels, chunked by 128 cols
  for (int c0 = 0; c0 < 1024; c0 += 128) {
    stage_w5_chunk(W + c0, wstage, wid, tid);
    for (int t = wid; t < 16; t += 8) {
      int rt = t >> 3, ct = t & 7;
      v8f acc = {};
#pragma unroll
      for (int k0 = 0; k0 < 128; k0 += 32) {
        v16bf A = load_A_lds(&abuf[rt * 16][k0], lane);
        v16bf B = load_B_ldsRM(wstage + (size_t)k0 * 128 + ct * 16, lane);
        acc = __builtin_amdgcn_wmma_f32_16x16x32_bf16(false, A, false, B, (short)0,
                                                      acc, false, false);
      }
      int col   = c0 + ct * 16 + (lane & 15);
      int rbase = rt * 16 + ((lane & 16) ? 8 : 0);
      float bv  = bias[col];
#pragma unroll
      for (int r = 0; r < 8; ++r) {
        float z = acc[r] + bv;
        atomicAdd(&rs[rbase + r], z);
        atomicAdd(&rss[rbase + r], z * z);
      }
    }
    __syncthreads();                           // wstage reused next chunk
  }
  if (tid < 32) {
    float mean = rs[tid] * (1.f / 1024.f);
    float var  = rss[tid] * (1.f / 1024.f) - mean * mean;
    rmean[tid] = mean;
    rrstd[tid] = rsqrtf(var + 1e-3f);
  }
  __syncthreads();
  // pass 2: recompute, LN+ReLU, per-channel max, chunked by 128 cols
  for (int c0 = 0; c0 < 1024; c0 += 128) {
    stage_w5_chunk(W + c0, wstage, wid, tid);
    for (int t = wid; t < 16; t += 8) {
      int rt = t >> 3, ct = t & 7;
      v8f acc = {};
#pragma unroll
      for (int k0 = 0; k0 < 128; k0 += 32) {
        v16bf A = load_A_lds(&abuf[rt * 16][k0], lane);
        v16bf B = load_B_ldsRM(wstage + (size_t)k0 * 128 + ct * 16, lane);
        acc = __builtin_amdgcn_wmma_f32_16x16x32_bf16(false, A, false, B, (short)0,
                                                      acc, false, false);
      }
      int col   = c0 + ct * 16 + (lane & 15);
      int lcol  = ct * 16 + (lane & 15);
      int rbase = rt * 16 + ((lane & 16) ? 8 : 0);
      float bv = bias[col], gg = g[col], ee = be[col];
#pragma unroll
      for (int r = 0; r < 8; ++r) {
        int row = rbase + r;
        float z = acc[r] + bv;
        z = (z - rmean[row]) * rrstd[row] * gg + ee;
        zbuf[row][lcol] = z > 0.f ? z : 0.f;
      }
    }
    __syncthreads();
    if (tid < 128) {
      float m = 0.f;
#pragma unroll 4
      for (int r = 0; r < 32; ++r) m = fmaxf(m, zbuf[r][tid]);
      atomicMax((unsigned int*)&outmax[c0 + tid], __float_as_uint(m));
    }
    __syncthreads();
  }
}

// ---------------------------------------------------------------------------
// Fused head kernel: 32 points per workgroup (256 thr / 8 waves).
// group = point_index / 512 selects the max-reduction row.
// ---------------------------------------------------------------------------
__global__ __launch_bounds__(256)
void k_head_conv(const float* __restrict__ X,
                 const unsigned short* W1, const float* b1, const float* g1, const float* e1,
                 const unsigned short* W2, const float* b2, const float* g2, const float* e2,
                 const unsigned short* W3, const float* b3, const float* g3, const float* e3,
                 const unsigned short* W4, const float* b4, const float* g4, const float* e4,
                 const unsigned short* W5, const float* b5, const float* g5, const float* e5,
                 float* __restrict__ outmax) {
  __shared__ unsigned short abuf[32][ASTR];
  __shared__ float zbuf[32][ZSTR];
  __shared__ __align__(16) unsigned short wshare[128 * 128];  // wT (stride 72) / TDM chunk
  __shared__ float rs[32], rss[32], rmean[32], rrstd[32];

  int tid = threadIdx.x, wid = tid >> 5, lane = tid & 31;
  int pt0 = blockIdx.x * 32;
  int group = pt0 >> 9;                      // / 512 points per group

  // load 32 points x 3 coords, zero-pad K to 32
  for (int i = tid; i < 32 * 32; i += 256) {
    int r = i >> 5, k = i & 31;
    float v = (k < 3) ? X[(size_t)(pt0 + r) * 3 + k] : 0.f;
    abuf[r][k] = f2bf(v);
  }
  __syncthreads();

  conv_layer<32, 64>(abuf, zbuf, wshare, W1, b1, g1, e1, wid, lane, tid);   // 3(pad) -> 64
  conv_layer<64, 64>(abuf, zbuf, wshare, W2, b2, g2, e2, wid, lane, tid);   // 64 -> 64
  conv_layer<64, 64>(abuf, zbuf, wshare, W3, b3, g3, e3, wid, lane, tid);   // 64 -> 64
  conv_layer<64, 128>(abuf, zbuf, wshare, W4, b4, g4, e4, wid, lane, tid);  // 64 -> 128
  conv_final(abuf, zbuf, wshare, W5, b5, g5, e5, outmax + (size_t)group * 1024,
             wid, lane, tid, rmean, rrstd, rs, rss);                        // 128 -> 1024 + max
}

// ---------------------------------------------------------------------------
// WMMA GEMM for the dense head layers: Z[M,N] = A[M,K] @ W[K,N] + b
// one wave per 16x16 output tile.
// ---------------------------------------------------------------------------
__global__ __launch_bounds__(32)
void k_gemm_bias(const float* __restrict__ A, const unsigned short* __restrict__ W,
                 const float* __restrict__ bias, float* __restrict__ Z, int K, int N) {
  int lane = threadIdx.x;
  int row0 = blockIdx.x * 16;
  int col0 = blockIdx.y * 16;
  v8f acc = {};
  for (int k0 = 0; k0 < K; k0 += 32) {
    v16bf a = load_A_gf32(A + (size_t)row0 * K, K, k0, lane);
    v16bf b = load_B_g(W + (size_t)k0 * N + col0, N, lane);
    acc = __builtin_amdgcn_wmma_f32_16x16x32_bf16(false, a, false, b, (short)0,
                                                  acc, false, false);
  }
  int col   = col0 + (lane & 15);
  int rbase = row0 + ((lane & 16) ? 8 : 0);
  float bv  = bias[col];
#pragma unroll
  for (int r = 0; r < 8; ++r) Z[(size_t)(rbase + r) * N + col] = acc[r] + bv;
}

// LayerNorm + ReLU over rows (in place)
__global__ __launch_bounds__(256)
void k_ln_relu(float* __restrict__ X, const float* __restrict__ g,
               const float* __restrict__ be, int N) {
  int tid = threadIdx.x;
  float* x = X + (size_t)blockIdx.x * N;
  __shared__ float s1[256], s2[256];
  float s = 0.f, ss = 0.f;
  for (int c = tid; c < N; c += 256) { float v = x[c]; s += v; ss += v * v; }
  s1[tid] = s; s2[tid] = ss; __syncthreads();
  for (int o = 128; o > 0; o >>= 1) {
    if (tid < o) { s1[tid] += s1[tid + o]; s2[tid] += s2[tid + o]; }
    __syncthreads();
  }
  float mean = s1[0] / N;
  float rstd = rsqrtf(s2[0] / N - mean * mean + 1e-3f);
  for (int c = tid; c < N; c += 256) {
    float v = (x[c] - mean) * rstd * g[c] + be[c];
    x[c] = v > 0.f ? v : 0.f;
  }
}

// plain f32 GEMM + bias (tiny layers): one block per row
__global__ void k_plain_gemm(const float* __restrict__ A, const float* __restrict__ W,
                             const float* __restrict__ b, float* __restrict__ out,
                             int K, int N) {
  int row = blockIdx.x;
  for (int c = threadIdx.x; c < N; c += blockDim.x) {
    float s = b[c];
    for (int k = 0; k < K; ++k) s += A[(size_t)row * K + k] * W[(size_t)k * N + c];
    out[(size_t)row * N + c] = s;
  }
}

// decoder hidden: dense + LN + ReLU, one block (<=64 thr) per row, N <= 36
__global__ __launch_bounds__(64)
void k_dense_ln_row(const float* __restrict__ in, const float* __restrict__ W,
                    const float* __restrict__ b, const float* __restrict__ g,
                    const float* __restrict__ be, float* __restrict__ out,
                    int K, int N) {
  int row = blockIdx.x, c = threadIdx.x;
  __shared__ float z[64];
  __shared__ float st[2];
  if (c < N) {
    float s = b[c];
    for (int k = 0; k < K; ++k) s += in[(size_t)row * K + k] * W[(size_t)k * N + c];
    z[c] = s;
  }
  __syncthreads();
  if (c == 0) {
    float s = 0.f, ss = 0.f;
    for (int i = 0; i < N; ++i) { s += z[i]; ss += z[i] * z[i]; }
    float mean = s / N;
    st[0] = mean; st[1] = rsqrtf(ss / N - mean * mean + 1e-3f);
  }
  __syncthreads();
  if (c < N) {
    float v = (z[c] - st[0]) * st[1] * g[c] + be[c];
    out[(size_t)row * N + c] = v > 0.f ? v : 0.f;
  }
}

// q = transfm / ||transfm||  (B=32 rows of 4)
__global__ void k_qnorm(const float* __restrict__ t, float* __restrict__ q) {
  int b = threadIdx.x;
  if (b >= 32) return;
  float x = t[b * 4], y = t[b * 4 + 1], z = t[b * 4 + 2], w = t[b * 4 + 3];
  float inv = rsqrtf(x * x + y * y + z * z + w * w);
  q[b * 4] = x * inv; q[b * 4 + 1] = y * inv; q[b * 4 + 2] = z * inv; q[b * 4 + 3] = w * inv;
}

// pc_rot[r,b,n,:] = R[r] @ pc[b,n,:]
__global__ void k_rotate(const float* __restrict__ pc, const float* __restrict__ Rm,
                         float* __restrict__ out) {
  int idx = blockIdx.x * blockDim.x + threadIdx.x;
  if (idx >= 4 * 32 * 512) return;
  int n = idx & 511, b = (idx >> 9) & 31, r = idx >> 14;
  const float* p = pc + (size_t)(b * 512 + n) * 3;
  const float* M = Rm + r * 9;
  float p0 = p[0], p1 = p[1], p2 = p[2];
  out[(size_t)idx * 3]     = M[0] * p0 + M[1] * p1 + M[2] * p2;
  out[(size_t)idx * 3 + 1] = M[3] * p0 + M[4] * p1 + M[5] * p2;
  out[(size_t)idx * 3 + 2] = M[6] * p0 + M[7] * p1 + M[8] * p2;
}

// rotated = p + w*t + cross(v,t),  t = 2*cross(v,p),  q indexed by batch b
__global__ void k_quat(const float* __restrict__ dec, const float* __restrict__ q,
                       float* __restrict__ out) {
  int idx = blockIdx.x * blockDim.x + threadIdx.x;
  if (idx >= 4 * 32 * 512) return;
  int b = (idx >> 9) & 31;
  float px = dec[(size_t)idx * 3], py = dec[(size_t)idx * 3 + 1], pz = dec[(size_t)idx * 3 + 2];
  const float* Q = q + b * 4;
  float vx = Q[0], vy = Q[1], vz = Q[2], w = Q[3];
  float tx = 2.f * (vy * pz - vz * py);
  float ty = 2.f * (vz * px - vx * pz);
  float tz = 2.f * (vx * py - vy * px);
  out[(size_t)idx * 3]     = px + w * tx + (vy * tz - vz * ty);
  out[(size_t)idx * 3 + 1] = py + w * ty + (vz * tx - vx * tz);
  out[(size_t)idx * 3 + 2] = pz + w * tz + (vx * ty - vy * tx);
}

// crc loss: sum over 6 (r1<r2) pairs x B of ||enc[r1,b]-enc[r2,b]||^2
__global__ __launch_bounds__(128)
void k_crc(const float* __restrict__ enc, float* __restrict__ acc) {
  static const int P1[6] = {0, 0, 0, 1, 1, 2};
  static const int P2[6] = {1, 2, 3, 2, 3, 3};
  int pair = blockIdx.x >> 5, b = blockIdx.x & 31;
  int r1 = P1[pair], r2 = P2[pair];
  int c = threadIdx.x;
  float d = enc[(size_t)(r1 * 32 + b) * 128 + c] - enc[(size_t)(r2 * 32 + b) * 128 + c];
  __shared__ float red[128];
  red[c] = d * d; __syncthreads();
  for (int o = 64; o > 0; o >>= 1) {
    if (c < o) red[c] += red[c + o];
    __syncthreads();
  }
  if (c == 0) atomicAdd(&acc[0], red[0]);
}

// overlap loss: per (r,b), mindis_i = min(min_{j!=i} d_ij, 1.0); sum (1 - min(mindis,1))
__global__ __launch_bounds__(256)
void k_overlap(const float* __restrict__ rot, float* __restrict__ acc) {
  int rb = blockIdx.x;
  __shared__ float px[512], py[512], pz[512];
  const float* p = rot + (size_t)rb * 512 * 3;
  for (int i = threadIdx.x; i < 512; i += 256) {
    px[i] = p[(size_t)i * 3]; py[i] = p[(size_t)i * 3 + 1]; pz[i] = p[(size_t)i * 3 + 2];
  }
  __syncthreads();
  float sum = 0.f;
  for (int i = threadIdx.x; i < 512; i += 256) {
    float mind = 3.4e38f, xi = px[i], yi = py[i], zi = pz[i];
    for (int j = 0; j < 512; ++j) {
      float dx = xi - px[j], dy = yi - py[j], dz = zi - pz[j];
      float d = dx * dx + dy * dy + dz * dz;
      if (j != i) mind = fminf(mind, d);
    }
    sum += 1.0f - fminf(mind, 1.0f);
  }
  __shared__ float red[256];
  red[threadIdx.x] = sum; __syncthreads();
  for (int o = 128; o > 0; o >>= 1) {
    if (threadIdx.x < o) red[threadIdx.x] += red[threadIdx.x + o];
    __syncthreads();
  }
  if (threadIdx.x == 0) atomicAdd(&acc[1], red[0]);
}

__global__ void k_finalize(const float* __restrict__ acc, float* __restrict__ out) {
  if (threadIdx.x == 0 && blockIdx.x == 0) {
    out[196608] = acc[0] / 192.0f;      // 6 pairs * 32 batches
    out[196609] = acc[1] / 262144.0f;
  }
}

__global__ void k_zero(float* p, int n) {
  int i = blockIdx.x * blockDim.x + threadIdx.x;
  if (i < n) p[i] = 0.f;
}

__global__ void k_cvt(const float* __restrict__ s, unsigned short* __restrict__ d, int n) {
  int i = blockIdx.x * blockDim.x + threadIdx.x;
  if (i < n) d[i] = f2bf(s[i]);
}

// zero-pad K -> Kp rows while converting (for the 3x64 first-layer weight)
__global__ void k_cvt_pad(const float* __restrict__ s, unsigned short* __restrict__ d,
                          int K, int Kp, int N) {
  int i = blockIdx.x * blockDim.x + threadIdx.x;
  if (i >= Kp * N) return;
  int k = i / N, c = i - k * N;
  d[i] = (k < K) ? f2bf(s[k * N + c]) : (unsigned short)0;
}

// ---------------------------------------------------------------------------
extern "C" void kernel_launch(void* const* d_in, const int* in_sizes, int n_in,
                              void* d_out, int out_size, void* d_ws, size_t ws_size,
                              hipStream_t stream) {
  (void)in_sizes; (void)n_in; (void)out_size; (void)ws_size;

  const float* pc  = (const float*)d_in[0];
  const float* rot = (const float*)d_in[1];
  const float *trW[5], *trB[5], *trG[5], *trE[5];
  const float *enW[5], *enB[5], *enG[5], *enE[5];
  for (int i = 0; i < 5; ++i) {
    trW[i] = (const float*)d_in[2 + 4 * i];  trB[i] = (const float*)d_in[3 + 4 * i];
    trG[i] = (const float*)d_in[4 + 4 * i];  trE[i] = (const float*)d_in[5 + 4 * i];
    enW[i] = (const float*)d_in[32 + 4 * i]; enB[i] = (const float*)d_in[33 + 4 * i];
    enG[i] = (const float*)d_in[34 + 4 * i]; enE[i] = (const float*)d_in[35 + 4 * i];
  }
  const float *trDW[2], *trDB[2], *trDG[2], *trDE[2];
  const float *enDW[2], *enDB[2], *enDG[2], *enDE[2];
  for (int j = 0; j < 2; ++j) {
    trDW[j] = (const float*)d_in[22 + 4 * j]; trDB[j] = (const float*)d_in[23 + 4 * j];
    trDG[j] = (const float*)d_in[24 + 4 * j]; trDE[j] = (const float*)d_in[25 + 4 * j];
    enDW[j] = (const float*)d_in[52 + 4 * j]; enDB[j] = (const float*)d_in[53 + 4 * j];
    enDG[j] = (const float*)d_in[54 + 4 * j]; enDE[j] = (const float*)d_in[55 + 4 * j];
  }
  const float* trFW = (const float*)d_in[30];  const float* trFB = (const float*)d_in[31];
  const float* enFW = (const float*)d_in[60];  const float* enFB = (const float*)d_in[61];
  const float *dhW[3], *dhB[3], *dhG[3], *dhE[3];
  for (int i = 0; i < 3; ++i) {
    dhW[i] = (const float*)d_in[62 + 4 * i]; dhB[i] = (const float*)d_in[63 + 4 * i];
    dhG[i] = (const float*)d_in[64 + 4 * i]; dhE[i] = (const float*)d_in[65 + 4 * i];
  }
  const float* dfW = (const float*)d_in[74];  const float* dfB = (const float*)d_in[75];

  // ---- workspace layout ----
  char* w = (char*)d_ws;
  auto alloc = [&](size_t bytes) -> char* {
    char* p = w; w += (bytes + 255) & ~(size_t)255; return p;
  };
  float* pcrot   = (float*)alloc((size_t)196608 * 4);
  float* trmax   = (float*)alloc((size_t)32 * 1024 * 4);
  float* encmax  = (float*)alloc((size_t)128 * 1024 * 4);
  float* t0      = (float*)alloc((size_t)128 * 512 * 4);
  float* t1      = (float*)alloc((size_t)128 * 256 * 4);
  float* transfm = (float*)alloc(32 * 4 * 4);
  float* qbuf    = (float*)alloc(32 * 4 * 4);
  float* encz    = (float*)alloc((size_t)128 * 128 * 4);
  float* h0      = (float*)alloc((size_t)128 * 64 * 4);
  float* h1      = (float*)alloc((size_t)128 * 64 * 4);
  float* decout  = (float*)alloc((size_t)128 * 1536 * 4);
  float* lacc    = (float*)alloc(256);
  unsigned short* c_trW1 = (unsigned short*)alloc((size_t)32 * 64 * 2);
  unsigned short* c_trW2 = (unsigned short*)alloc((size_t)64 * 64 * 2);
  unsigned short* c_trW3 = (unsigned short*)alloc((size_t)64 * 64 * 2);
  unsigned short* c_trW4 = (unsigned short*)alloc((size_t)64 * 128 * 2);
  unsigned short* c_trW5 = (unsigned short*)alloc((size_t)128 * 1024 * 2);
  unsigned short* c_enW1 = (unsigned short*)alloc((size_t)32 * 64 * 2);
  unsigned short* c_enW2 = (unsigned short*)alloc((size_t)64 * 64 * 2);
  unsigned short* c_enW3 = (unsigned short*)alloc((size_t)64 * 64 * 2);
  unsigned short* c_enW4 = (unsigned short*)alloc((size_t)64 * 128 * 2);
  unsigned short* c_enW5 = (unsigned short*)alloc((size_t)128 * 1024 * 2);
  unsigned short* c_trD1 = (unsigned short*)alloc((size_t)1024 * 512 * 2);
  unsigned short* c_trD2 = (unsigned short*)alloc((size_t)512 * 256 * 2);
  unsigned short* c_enD1 = (unsigned short*)alloc((size_t)1024 * 512 * 2);
  unsigned short* c_enD2 = (unsigned short*)alloc((size_t)512 * 256 * 2);

  auto nb = [](int n, int t) { return (n + t - 1) / t; };

  // ---- init accumulators (post-ReLU max >= 0 => zero-init is correct) ----
  k_zero<<<nb(32 * 1024, 256), 256, 0, stream>>>(trmax, 32 * 1024);
  k_zero<<<nb(128 * 1024, 256), 256, 0, stream>>>(encmax, 128 * 1024);
  k_zero<<<1, 256, 0, stream>>>(lacc, 2);

  // ---- convert weights to bf16 (W1 zero-padded 3->32 rows) ----
  k_cvt_pad<<<nb(32 * 64, 256), 256, 0, stream>>>(trW[0], c_trW1, 3, 32, 64);
  k_cvt<<<nb(4096, 256), 256, 0, stream>>>(trW[1], c_trW2, 4096);
  k_cvt<<<nb(4096, 256), 256, 0, stream>>>(trW[2], c_trW3, 4096);
  k_cvt<<<nb(8192, 256), 256, 0, stream>>>(trW[3], c_trW4, 8192);
  k_cvt<<<nb(131072, 256), 256, 0, stream>>>(trW[4], c_trW5, 131072);
  k_cvt_pad<<<nb(32 * 64, 256), 256, 0, stream>>>(enW[0], c_enW1, 3, 32, 64);
  k_cvt<<<nb(4096, 256), 256, 0, stream>>>(enW[1], c_enW2, 4096);
  k_cvt<<<nb(4096, 256), 256, 0, stream>>>(enW[2], c_enW3, 4096);
  k_cvt<<<nb(8192, 256), 256, 0, stream>>>(enW[3], c_enW4, 8192);
  k_cvt<<<nb(131072, 256), 256, 0, stream>>>(enW[4], c_enW5, 131072);
  k_cvt<<<nb(524288, 256), 256, 0, stream>>>(trDW[0], c_trD1, 524288);
  k_cvt<<<nb(131072, 256), 256, 0, stream>>>(trDW[1], c_trD2, 131072);
  k_cvt<<<nb(524288, 256), 256, 0, stream>>>(enDW[0], c_enD1, 524288);
  k_cvt<<<nb(131072, 256), 256, 0, stream>>>(enDW[1], c_enD2, 131072);

  // ---- rotated point clouds ----
  k_rotate<<<nb(65536, 256), 256, 0, stream>>>(pc, rot, pcrot);

  // ---- fused WMMA conv heads (+ max over N) ----
  k_head_conv<<<512, 256, 0, stream>>>(pc,
      c_trW1, trB[0], trG[0], trE[0],  c_trW2, trB[1], trG[1], trE[1],
      c_trW3, trB[2], trG[2], trE[2],  c_trW4, trB[3], trG[3], trE[3],
      c_trW5, trB[4], trG[4], trE[4],  trmax);
  k_head_conv<<<2048, 256, 0, stream>>>(pcrot,
      c_enW1, enB[0], enG[0], enE[0],  c_enW2, enB[1], enG[1], enE[1],
      c_enW3, enB[2], enG[2], enE[2],  c_enW4, enB[3], enG[3], enE[3],
      c_enW5, enB[4], enG[4], enE[4],  encmax);

  // ---- transform head denses (WMMA) -> quaternion ----
  k_gemm_bias<<<dim3(2, 32), 32, 0, stream>>>(trmax, c_trD1, trDB[0], t0, 1024, 512);
  k_ln_relu<<<32, 256, 0, stream>>>(t0, trDG[0], trDE[0], 512);
  k_gemm_bias<<<dim3(2, 16), 32, 0, stream>>>(t0, c_trD2, trDB[1], t1, 512, 256);
  k_ln_relu<<<32, 256, 0, stream>>>(t1, trDG[1], trDE[1], 256);
  k_plain_gemm<<<32, 64, 0, stream>>>(t1, trFW, trFB, transfm, 256, 4);
  k_qnorm<<<1, 32, 0, stream>>>(transfm, qbuf);

  // ---- encoder denses (WMMA) -> latent ----
  k_gemm_bias<<<dim3(8, 32), 32, 0, stream>>>(encmax, c_enD1, enDB[0], t0, 1024, 512);
  k_ln_relu<<<128, 256, 0, stream>>>(t0, enDG[0], enDE[0], 512);
  k_gemm_bias<<<dim3(8, 16), 32, 0, stream>>>(t0, c_enD2, enDB[1], t1, 512, 256);
  k_ln_relu<<<128, 256, 0, stream>>>(t1, enDG[1], enDE[1], 256);
  k_plain_gemm<<<128, 128, 0, stream>>>(t1, enFW, enFB, encz, 256, 128);

  // ---- decoder ----
  k_dense_ln_row<<<128, 64, 0, stream>>>(encz, dhW[0], dhB[0], dhG[0], dhE[0], h0, 128, 16);
  k_dense_ln_row<<<128, 64, 0, stream>>>(h0, dhW[1], dhB[1], dhG[1], dhE[1], h1, 16, 32);
  k_dense_ln_row<<<128, 64, 0, stream>>>(h1, dhW[2], dhB[2], dhG[2], dhE[2], h0, 32, 36);
  k_plain_gemm<<<128, 256, 0, stream>>>(h0, dfW, dfB, decout, 36, 1536);

  // ---- quaternion rotate -> d_out, then losses ----
  float* out = (float*)d_out;
  k_quat<<<nb(65536, 256), 256, 0, stream>>>(decout, qbuf, out);
  k_crc<<<192, 128, 0, stream>>>(encz, lacc);
  k_overlap<<<128, 256, 0, stream>>>(out, lacc);
  k_finalize<<<1, 1, 0, stream>>>(lacc, out);
}
